// LoRALayer_13056700580108
// MI455X (gfx1250) — matrix operand order repaired and checked by reading, not compiled, not run on hardware
//
#include <hip/hip_runtime.h>

// ---------------------------------------------------------------------------
// LoRA forward for MI455X (gfx1250, wave32, WMMA).
//   out = x @ W^T + b + 2.0 * (x @ B) @ A
// M=8192, N=4096, K=4096 -> compute bound (AI ~1700 FLOP/B) -> bf16 WMMA,
// fp32 accumulate. x/W pre-converted to bf16 in workspace (96MB working set
// stays L2-resident, 192MB L2). LoRA fused into epilogue via T = 2*(x@B).
// Tile fills use GLOBAL_LOAD_ASYNC_TO_LDS_B128 (ASYNCcnt DMA) with LDS
// double buffering so tile i+1's DMA overlaps tile i's WMMAs.
// ---------------------------------------------------------------------------

#define TOKENS 8192
#define DIMN   4096
#define KDIM   4096
#define RANK   16
#define LORA_SCALE 2.0f   // alpha/rank = 32/16

typedef float  v8f   __attribute__((ext_vector_type(8)));
typedef __bf16 v16bf __attribute__((ext_vector_type(16)));

union FragBF {
  v16bf v;
  uint4 q[2];
};

__device__ __forceinline__ unsigned int f2bf1(float f) {
  unsigned int u = __float_as_uint(f);
  return (u + 0x7FFFu + ((u >> 16) & 1u)) >> 16;  // RNE
}
__device__ __forceinline__ unsigned int packbf2(float lo, float hi) {
  return f2bf1(lo) | (f2bf1(hi) << 16);
}

// ---------------------------------------------------------------------------
// fp32 -> bf16 bulk conversion (grid sized exactly; 8 elems/thread).
// ---------------------------------------------------------------------------
__global__ __launch_bounds__(256) void cvt_f32_bf16(
    const float* __restrict__ in, unsigned short* __restrict__ out_) {
  size_t i = ((size_t)blockIdx.x * 256 + threadIdx.x) * 8;
  float4 a = *(const float4*)(in + i);
  float4 c = *(const float4*)(in + i + 4);
  uint4 p = make_uint4(packbf2(a.x, a.y), packbf2(a.z, a.w),
                       packbf2(c.x, c.y), packbf2(c.z, c.w));
  *(uint4*)(out_ + i) = p;
}

// ---------------------------------------------------------------------------
// T[t][r] = 2.0 * sum_k x[t,k] * B[k,r]   (B is [4096,16] row-major).
// ---------------------------------------------------------------------------
__global__ __launch_bounds__(256) void lora_xB(
    const float* __restrict__ x, const float* __restrict__ Bm,
    float* __restrict__ T) {
  int idx = blockIdx.x * 256 + threadIdx.x;   // 0 .. TOKENS*RANK-1
  int t = idx >> 4;
  int r = idx & 15;
  const float* xr = x + (size_t)t * KDIM;
  float acc = 0.f;
#pragma unroll 4
  for (int k = 0; k < KDIM; ++k) acc += xr[k] * Bm[k * RANK + r];
  T[idx] = acc * LORA_SCALE;
}

// ---------------------------------------------------------------------------
// Shared WMMA inner step: 2(M) x 4(N) subtiles of 16x16, K=32, per wave.
// A frag (16x32 bf16): lane l -> row l&15; lanes 0-15 hold K=0..7,16..23,
// lanes 16-31 K=8..15,24..31 (ISA 7.12.2) -> two b128 LDS reads.
// B frag (32x16 bf16): lane l -> col l&15, same K split as A -> two b128 reads.
// ---------------------------------------------------------------------------
__device__ __forceinline__ void mma_step(
    const unsigned short* __restrict__ Xs, const unsigned short* __restrict__ Wt,
    int wm, int wn, int hl, int kg, v8f acc[2][4]) {
  constexpr int LDK = 40;
  FragBF af[2];
#pragma unroll
  for (int ms = 0; ms < 2; ++ms) {
    const int base = (wm + ms * 16 + hl) * LDK + kg * 8;
    af[ms].q[0] = *(const uint4*)&Xs[base];
    af[ms].q[1] = *(const uint4*)&Xs[base + 16];
  }
  FragBF bq[4];
#pragma unroll
  for (int ns = 0; ns < 4; ++ns) {
    const int base = (wn + ns * 16 + hl) * LDK + kg * 8;
    bq[ns].q[0] = *(const uint4*)&Wt[base];
    bq[ns].q[1] = *(const uint4*)&Wt[base + 16];
  }
#pragma unroll
  for (int ms = 0; ms < 2; ++ms)
#pragma unroll
    for (int ns = 0; ns < 4; ++ns)
      acc[ms][ns] = __builtin_amdgcn_wmma_f32_16x16x32_bf16(
          false, af[ms].v, false, bq[ns].v, (short)0, acc[ms][ns],
          false, false);
}

// Epilogue: out = acc + bias + T(128x16) @ A(16x128), all LDS-resident.
__device__ __forceinline__ void epilogue(
    const float* __restrict__ Ts, const float* __restrict__ As,
    const float* __restrict__ Bs, v8f acc[2][4], int wm, int wn, int hl,
    int kg, int bm, int bn, float* __restrict__ out) {
#pragma unroll
  for (int ms = 0; ms < 2; ++ms) {
#pragma unroll
    for (int ns = 0; ns < 4; ++ns) {
      const int ml0 = wm + ms * 16 + kg * 8;  // C layout: M=(l>>4)*8+v
      const int nl  = wn + ns * 16 + hl;      //           N=l&15
      const float bv = Bs[nl];
#pragma unroll
      for (int v = 0; v < 8; ++v) {
        const int ml = ml0 + v;
        float val = acc[ms][ns][v] + bv;
#pragma unroll
        for (int r = 0; r < RANK; ++r)
          val += Ts[ml * RANK + r] * As[r * 128 + nl];
        out[(size_t)(bm + ml) * DIMN + bn + nl] = val;
      }
    }
  }
}

// ---------------------------------------------------------------------------
// Main GEMM, bf16 inputs, async-to-LDS double-buffered pipeline.
// Block tile 128x128, BK=32, 256 threads = 8 waves (4M x 2N).
// Per tile per wave: 4 global_load_async_to_lds_b128 (ASYNCcnt); in-order
// completion => s_wait_asynccnt 4 after issuing the next tile's 4 guarantees
// the current tile's DMA has landed.
// ---------------------------------------------------------------------------
__global__ __launch_bounds__(256) void lora_wmma_gemm_async(
    const unsigned short* __restrict__ xb, const unsigned short* __restrict__ wb,
    const float* __restrict__ bias, const float* __restrict__ A,
    const float* __restrict__ T, float* __restrict__ out) {
  constexpr int BM = 128, BN = 128, BK = 32, LDK = 40;  // 80B padded rows
  constexpr int KSTEPS = KDIM / BK;                     // 128
  __shared__ unsigned short Xs[2 * BM * LDK];
  __shared__ unsigned short Wt[2 * BN * LDK];
  __shared__ float Ts[BM * RANK];
  __shared__ float As[RANK * BN];
  __shared__ float Bs[BN];

  const int tid  = threadIdx.x;
  const int lane = tid & 31;
  const int wave = tid >> 5;
  const int wm = (wave >> 1) * 32;
  const int wn = (wave & 1) * 64;
  const int bm = blockIdx.y * BM;
  const int bn = blockIdx.x * BN;
  const int hl = lane & 15;
  const int kg = lane >> 4;

  // Epilogue operands (first barrier covers visibility).
  for (int i = tid; i < BM * RANK; i += 256) Ts[i] = T[bm * RANK + i];
  for (int i = tid; i < RANK * BN; i += 256)
    As[i] = A[(i >> 7) * KDIM + bn + (i & 127)];
  if (tid < BN) Bs[tid] = bias[bn + tid];

  const v8f vzero = {0.f, 0.f, 0.f, 0.f, 0.f, 0.f, 0.f, 0.f};
  v8f acc[2][4];
#pragma unroll
  for (int i = 0; i < 2; ++i)
#pragma unroll
    for (int j = 0; j < 4; ++j) acc[i][j] = vzero;

  // Fill mapping: each thread owns 32B (16 bf16) of one tile row.
  const int frow = tid >> 1;
  const int fcol = (tid & 1) * 16;
  const unsigned short* xrow = xb + (size_t)(bm + frow) * KDIM + fcol;
  const unsigned short* wrow = wb + (size_t)(bn + frow) * KDIM + fcol;
  // LDS byte offsets (generic-pointer truncation == wave LDS offset).
  const unsigned xl = (unsigned)(size_t)&Xs[frow * LDK + fcol];
  const unsigned wl = (unsigned)(size_t)&Wt[frow * LDK + fcol];
  constexpr unsigned stageBytes = BM * LDK * 2u;

  // INST_OFFSET applies to BOTH the LDS and global address (ISA 15.18.3),
  // so the second 16B chunk is just "offset:16" on the same base VGPRs.
  auto issue = [&](int kk, int stage) {
    const unsigned xo = xl + (unsigned)stage * stageBytes;
    const unsigned wo = wl + (unsigned)stage * stageBytes;
    const unsigned short* xg = xrow + kk;
    const unsigned short* wg = wrow + kk;
    asm volatile(
        "global_load_async_to_lds_b128 %0, %2, off\n\t"
        "global_load_async_to_lds_b128 %0, %2, off offset:16\n\t"
        "global_load_async_to_lds_b128 %1, %3, off\n\t"
        "global_load_async_to_lds_b128 %1, %3, off offset:16"
        :: "v"(xo), "v"(wo), "v"(xg), "v"(wg)
        : "memory");
  };

  issue(0, 0);  // prologue: tile 0 -> buffer 0

  for (int i = 0; i < KSTEPS; ++i) {
    // Prefetch next tile into the other buffer (wrap on last iter: benign
    // reload of tile 0 into a buffer nobody reads again; no OOB, no branch).
    const int nk = ((i + 1) & (KSTEPS - 1)) * BK;
    issue(nk, (i + 1) & 1);
    // 8 async loads may be outstanding; <=4 left => tile i's DMA landed.
    asm volatile("s_wait_asynccnt 0x4" ::: "memory");
    __syncthreads();
    mma_step(&Xs[(i & 1) * BM * LDK], &Wt[(i & 1) * BM * LDK],
             wm, wn, hl, kg, acc);
    __syncthreads();  // all waves done reading buf[i&1] before it is refilled
  }

  epilogue(Ts, As, Bs, acc, wm, wn, hl, kg, bm, bn, out);
}

// ---------------------------------------------------------------------------
// Fallback (workspace too small for bf16 copies): fp32 source, convert while
// filling LDS, single-buffered synchronous staging.
// ---------------------------------------------------------------------------
__global__ __launch_bounds__(256) void lora_wmma_gemm_f32src(
    const float* __restrict__ xf, const float* __restrict__ wf,
    const float* __restrict__ bias, const float* __restrict__ A,
    const float* __restrict__ T, float* __restrict__ out) {
  constexpr int BM = 128, BN = 128, BK = 32, LDK = 40;
  __shared__ unsigned short Xs[BM * LDK];
  __shared__ unsigned short Wt[BN * LDK];
  __shared__ float Ts[BM * RANK];
  __shared__ float As[RANK * BN];
  __shared__ float Bs[BN];

  const int tid  = threadIdx.x;
  const int lane = tid & 31;
  const int wave = tid >> 5;
  const int wm = (wave >> 1) * 32;
  const int wn = (wave & 1) * 64;
  const int bm = blockIdx.y * BM;
  const int bn = blockIdx.x * BN;
  const int hl = lane & 15;
  const int kg = lane >> 4;

  for (int i = tid; i < BM * RANK; i += 256) Ts[i] = T[bm * RANK + i];
  for (int i = tid; i < RANK * BN; i += 256)
    As[i] = A[(i >> 7) * KDIM + bn + (i & 127)];
  if (tid < BN) Bs[tid] = bias[bn + tid];

  const v8f vzero = {0.f, 0.f, 0.f, 0.f, 0.f, 0.f, 0.f, 0.f};
  v8f acc[2][4];
#pragma unroll
  for (int i = 0; i < 2; ++i)
#pragma unroll
    for (int j = 0; j < 4; ++j) acc[i][j] = vzero;

  const int frow = tid >> 1;
  const int fcol = (tid & 1) * 16;

  for (int kk = 0; kk < KDIM; kk += BK) {
    const size_t xo = (size_t)(bm + frow) * KDIM + kk + fcol;
    const size_t wo = (size_t)(bn + frow) * KDIM + kk + fcol;
    float4 a0 = *(const float4*)(xf + xo);
    float4 a1 = *(const float4*)(xf + xo + 4);
    float4 a2 = *(const float4*)(xf + xo + 8);
    float4 a3 = *(const float4*)(xf + xo + 12);
    float4 b0 = *(const float4*)(wf + wo);
    float4 b1 = *(const float4*)(wf + wo + 4);
    float4 b2 = *(const float4*)(wf + wo + 8);
    float4 b3 = *(const float4*)(wf + wo + 12);
    *(uint4*)&Xs[frow * LDK + fcol] =
        make_uint4(packbf2(a0.x, a0.y), packbf2(a0.z, a0.w),
                   packbf2(a1.x, a1.y), packbf2(a1.z, a1.w));
    *(uint4*)&Xs[frow * LDK + fcol + 8] =
        make_uint4(packbf2(a2.x, a2.y), packbf2(a2.z, a2.w),
                   packbf2(a3.x, a3.y), packbf2(a3.z, a3.w));
    *(uint4*)&Wt[frow * LDK + fcol] =
        make_uint4(packbf2(b0.x, b0.y), packbf2(b0.z, b0.w),
                   packbf2(b1.x, b1.y), packbf2(b1.z, b1.w));
    *(uint4*)&Wt[frow * LDK + fcol + 8] =
        make_uint4(packbf2(b2.x, b2.y), packbf2(b2.z, b2.w),
                   packbf2(b3.x, b3.y), packbf2(b3.z, b3.w));
    __syncthreads();
    mma_step(Xs, Wt, wm, wn, hl, kg, acc);
    __syncthreads();
  }

  epilogue(Ts, As, Bs, acc, wm, wn, hl, kg, bm, bn, out);
}

// ---------------------------------------------------------------------------
extern "C" void kernel_launch(void* const* d_in, const int* in_sizes, int n_in,
                              void* d_out, int out_size, void* d_ws,
                              size_t ws_size, hipStream_t stream) {
  const float* x    = (const float*)d_in[0];  // [8192, 4096]
  const float* W    = (const float*)d_in[1];  // [4096, 4096]
  const float* bias = (const float*)d_in[2];  // [4096]
  const float* A    = (const float*)d_in[3];  // [16, 4096]
  const float* Bm   = (const float*)d_in[4];  // [4096, 16]
  float* out = (float*)d_out;

  const size_t xbf_bytes = (size_t)TOKENS * KDIM * 2;  // 64 MB
  const size_t wbf_bytes = (size_t)DIMN * KDIM * 2;    // 32 MB
  const size_t t_bytes   = (size_t)TOKENS * RANK * 4;  // 512 KB
  const bool pre = ws_size >= xbf_bytes + wbf_bytes + t_bytes;

  char* ws = (char*)d_ws;
  unsigned short* xbf = (unsigned short*)ws;
  unsigned short* wbf = (unsigned short*)(ws + xbf_bytes);
  float* T = pre ? (float*)(ws + xbf_bytes + wbf_bytes) : (float*)ws;

  // LoRA intermediate: T = 2.0 * x @ B
  lora_xB<<<(TOKENS * RANK) / 256, 256, 0, stream>>>(x, Bm, T);

  dim3 grid(DIMN / 128, TOKENS / 128);  // 32 x 64 blocks
  if (pre) {
    cvt_f32_bf16<<<((size_t)TOKENS * KDIM) / (8 * 256), 256, 0, stream>>>(x, xbf);
    cvt_f32_bf16<<<((size_t)DIMN * KDIM) / (8 * 256), 256, 0, stream>>>(W, wbf);
    lora_wmma_gemm_async<<<grid, 256, 0, stream>>>(xbf, wbf, bias, A, T, out);
  } else {
    lora_wmma_gemm_f32src<<<grid, 256, 0, stream>>>(x, W, bias, A, T, out);
  }
}
